// HiddenStateAligner_67250597921155
// MI455X (gfx1250) — compile-verified, matrix operand order checked by reading, not run on hardware
//
#include <hip/hip_runtime.h>
#include <hip/hip_bf16.h>
#include <math.h>
#include <stdint.h>

// ---------------------------------------------------------------------------
// HiddenStateAligner for MI455X (gfx1250, wave32, WMMA)
//   K0: pack W_off / W_dcn once into im2col-ordered bf16 panels (workspace)
//   K1: offset conv  GEMM  M=112(pad of 108) x K=2304 x N=64px tiles
//   K2: deformable sample (L2-resident gathers) + DCN GEMM M=128 x K=1152
// WMMA: v_wmma_f32_16x16x32_bf16, fp32 accumulate.
// A-panels staged to LDS via global_load_async_to_lds_b128 when available.
// ---------------------------------------------------------------------------

typedef __attribute__((ext_vector_type(16))) __bf16 v16bf;
typedef __attribute__((ext_vector_type(8)))  float  v8f;
typedef int hsa_v4i __attribute__((__vector_size__(4 * sizeof(int))));

#define BATCH 4
#define CH    128
#define CF    128
#define GCH   256
#define HH    128
#define WW    128
#define HW    (HH * WW)
#define G     4
#define KK    9
#define CG    32
#define OFFC  108
#define OFFCP 112
#define MAXRES 3.0f
#define K1TOT 2304      // GCH * KK
#define K2TOT 1152      // G * CG * KK

#if __has_builtin(__builtin_amdgcn_global_load_async_to_lds_b128) && \
    __has_builtin(__builtin_amdgcn_s_wait_asynccnt)
#define HSA_ASYNC 1
#else
#define HSA_ASYNC 0
#endif

#if HSA_ASYNC
__device__ __forceinline__ void hsa_async_b128(const void* g, void* l) {
    __builtin_amdgcn_global_load_async_to_lds_b128(
        (__attribute__((address_space(1))) hsa_v4i*)(uintptr_t)g,
        (__attribute__((address_space(3))) hsa_v4i*)(uintptr_t)l,
        0, 0);
}
#endif

__device__ __forceinline__ float hsa_tanh(float x) {
#if __has_builtin(__builtin_amdgcn_tanhf)
    return __builtin_amdgcn_tanhf(x);          // V_TANH_F32 on gfx1250
#else
    const float e = __expf(-2.0f * x);         // tanh = 1 - 2e/(1+e)
    return 1.0f - 2.0f * e * __builtin_amdgcn_rcpf(1.0f + e);
#endif
}

__device__ __forceinline__ float hsa_sigmoid2(float x) {   // 2*sigmoid(x)
    return 2.0f * __builtin_amdgcn_rcpf(1.0f + __expf(-x));
}

// --- fragment loader: A/B 16-bit 16x32 layout --------------------------------
// lane<16 : K = {0..7, 16..23};  lane>=16 : K = {8..15, 24..31}; row = lane%16
__device__ __forceinline__ v16bf load_frag_row32(const __bf16* lds_row32, int lane) {
    const int k0 = (lane < 16) ? 0 : 8;
    v16bf f;
#pragma unroll
    for (int i = 0; i < 8; ++i) {
        f[i]     = lds_row32[k0 + i];
        f[i + 8] = lds_row32[k0 + 16 + i];
    }
    return f;
}

// ===========================================================================
// Kernel 0: pack weights to bf16 im2col panels (runs once, L2-resident)
//   A1[oc][tap*256 + c]       = W_off[oc, c, tap]        (112 x 2304)
//   A2[o][(g*9+k)*32 + c]     = W_dcn[o, g*32+c, k]      (128 x 1152)
// ===========================================================================
__global__ __launch_bounds__(256) void hsa_pack_weights(
    const float* __restrict__ W_off, const float* __restrict__ W_dcn,
    __bf16* __restrict__ A1, __bf16* __restrict__ A2)
{
    const int idx    = blockIdx.x * 256 + threadIdx.x;
    const int stride = gridDim.x * 256;
    for (int i = idx; i < OFFCP * K1TOT; i += stride) {
        const int oc = i / K1TOT, r = i % K1TOT;
        const int tap = r >> 8, c = r & 255;
        A1[i] = (__bf16)((oc < OFFC) ? W_off[oc * K1TOT + c * KK + tap] : 0.0f);
    }
    for (int i = idx; i < CH * K2TOT; i += stride) {
        const int o = i / K2TOT, r = i % K2TOT;
        const int q = r >> 5, c = r & 31;
        const int g = q / KK, k = q % KK;
        A2[i] = (__bf16)W_dcn[((size_t)o * CH + g * CG + c) * KK + k];
    }
}

// ===========================================================================
// Kernel 1: offset conv -> tanh/sigmoid -> dy, dx, mask
//   64-pixel tile per WG; 7 waves x (1 M-tile x 4 N-tiles) WMMAs per chunk
// ===========================================================================
__global__ __launch_bounds__(256) void hsa_offset_conv(
    const float* __restrict__ x_t, const float* __restrict__ h_prev,
    const __bf16* __restrict__ A1, const float* __restrict__ b_off,
    float* __restrict__ dy, float* __restrict__ dx, float* __restrict__ mask)
{
    __shared__ __bf16 At[OFFCP * 32];   // 7 KB
    __shared__ __bf16 Bt[64 * 32];      // 4 KB

    const int tid  = threadIdx.x;
    const int lane = tid & 31;
    const int wave = tid >> 5;

    const int bid = blockIdx.x;             // B * H * (W/64) = 1024
    const int b   = bid >> 8;
    const int rem = bid & 255;
    const int h   = rem >> 1;
    const int x0  = (rem & 1) << 6;

    v8f acc[4] = {};

    for (int q = 0; q < 72; ++q) {          // K chunks of 32 (tap-major)
        const int tap   = q >> 3;
        const int cbase = (q & 7) << 5;
        const int ky    = tap / 3 - 1;
        const int kx    = tap % 3 - 1;

        // ---- stage A panel chunk (pure 16B block copy) --------------------
        for (int i = tid; i < OFFCP * 4; i += 256) {
            const int row = i >> 2, seg = i & 3;
            const __bf16* gsrc = A1 + (size_t)row * K1TOT + q * 32 + seg * 8;
#if HSA_ASYNC
            hsa_async_b128(gsrc, &At[row * 32 + seg * 8]);
#else
            *(uint4*)&At[row * 32 + seg * 8] = *(const uint4*)gsrc;
#endif
        }

        // ---- stage B: guide patch, 64 px x 32 ch, zero-padded -------------
        {
            const int yy = h + ky;
            for (int e = tid; e < 64 * 32; e += 256) {
                const int p  = e >> 5;
                const int c  = e & 31;
                const int xx = x0 + p + kx;
                float v = 0.0f;
                if (yy >= 0 && yy < HH && xx >= 0 && xx < WW) {
                    const int cc = cbase + c;
                    v = (cc < CF)
                        ? x_t[((size_t)(b * CF + cc) * HH + yy) * WW + xx]
                        : h_prev[((size_t)(b * CH + (cc - CF)) * HH + yy) * WW + xx];
                }
                Bt[p * 32 + c] = (__bf16)v;
            }
        }
#if HSA_ASYNC
        __builtin_amdgcn_s_wait_asynccnt(0);
#endif
        __syncthreads();

        // ---- WMMA: 7 waves x 4 N-tiles ------------------------------------
        if (wave < 7) {
            const v16bf a = load_frag_row32(&At[(wave * 16 + (lane & 15)) * 32], lane);
#pragma unroll
            for (int nt = 0; nt < 4; ++nt) {
                const v16bf bb =
                    load_frag_row32(&Bt[(nt * 16 + (lane & 15)) * 32], lane);
                acc[nt] = __builtin_amdgcn_wmma_f32_16x16x32_bf16(
                    false, a, false, bb, (short)0, acc[nt], false, false);
            }
        }
        __syncthreads();
    }

    // ---- epilogue: bias + tanh/sigmoid, scatter to dy/dx/mask -------------
    if (wave < 7) {
        const int n = lane & 15;
#pragma unroll
        for (int nt = 0; nt < 4; ++nt) {
            const int x = x0 + nt * 16 + n;
#pragma unroll
            for (int r = 0; r < 8; ++r) {
                const int m = wave * 16 + r + ((lane >= 16) ? 8 : 0);
                if (m >= OFFC) continue;
                const float v = acc[nt][r] + b_off[m];
                if (m < 2 * KK * G) {
                    const int gk = m >> 1, d = m & 1;
                    const int g = gk / KK, k = gk % KK;
                    float* dst = (d == 0) ? dy : dx;
                    dst[(((size_t)(b * G + g) * KK + k) * HH + h) * WW + x] =
                        hsa_tanh(v) * MAXRES;
                } else {
                    const int mm = m - 2 * KK * G;
                    const int g = mm / KK, k = mm % KK;
                    mask[(((size_t)(b * G + g) * KK + k) * HH + h) * WW + x] =
                        hsa_sigmoid2(v);
                }
            }
        }
    }
}

// ===========================================================================
// Kernel 2: deformable bilinear sample + DCN GEMM
//   64-pixel tile per WG; 8 waves x (1 M-tile x 4 N-tiles) WMMAs per chunk
// ===========================================================================
__global__ __launch_bounds__(256) void hsa_dcn_sample_gemm(
    const float* __restrict__ h_prev, const __bf16* __restrict__ A2,
    const float* __restrict__ b_dcn,
    const float* __restrict__ dy, const float* __restrict__ dx,
    const float* __restrict__ maskp, float* __restrict__ out)
{
    __shared__ __bf16 At[CH * 32];   // 8 KB
    __shared__ __bf16 Bt[64 * 32];   // 4 KB
    __shared__ float  sdy[64], sdx[64], smk[64];

    const int tid  = threadIdx.x;
    const int lane = tid & 31;
    const int wave = tid >> 5;

    const int bid = blockIdx.x;
    const int b   = bid >> 8;
    const int rem = bid & 255;
    const int h   = rem >> 1;
    const int x0  = (rem & 1) << 6;

    v8f acc[4] = {};

    for (int q = 0; q < 36; ++q) {          // K chunks: (g,k) x 32 channels
        const int g  = q / KK;
        const int k  = q % KK;
        const int ky = k / 3 - 1;
        const int kx = k % 3 - 1;

        // ---- per-pixel offsets/mask + A panel chunk -----------------------
        if (tid < 64) {
            const size_t idx =
                (((size_t)(b * G + g) * KK + k) * HH + h) * WW + (x0 + tid);
            sdy[tid] = dy[idx];
            sdx[tid] = dx[idx];
            smk[tid] = maskp[idx];
        }
        for (int i = tid; i < CH * 4; i += 256) {
            const int row = i >> 2, seg = i & 3;
            const __bf16* gsrc = A2 + (size_t)row * K2TOT + q * 32 + seg * 8;
#if HSA_ASYNC
            hsa_async_b128(gsrc, &At[row * 32 + seg * 8]);
#else
            *(uint4*)&At[row * 32 + seg * 8] = *(const uint4*)gsrc;
#endif
        }
        __syncthreads();                    // sdy/sdx/smk visible

        // ---- bilinear sample: thread -> pixel (tid&63), 8 channels --------
        {
            const int p  = tid & 63;
            const int c0 = (tid >> 6) << 3;
            const float py = (float)h + (float)ky + sdy[p];
            const float px = (float)(x0 + p) + (float)kx + sdx[p];
            const float mk = smk[p];
            const float fy0 = floorf(py), fx0 = floorf(px);
            const float wy1 = py - fy0, wx1 = px - fx0;
            const float wy0 = 1.0f - wy1, wx0 = 1.0f - wx1;
            const int yi = (int)fy0;
            const int xi = (int)fx0;
            const bool y0ok = (yi >= 0) && (yi < HH);
            const bool y1ok = (yi + 1 >= 0) && (yi + 1 < HH);
            const bool x0ok = (xi >= 0) && (xi < WW);
            const bool x1ok = (xi + 1 >= 0) && (xi + 1 < WW);
            const float w00 = wy0 * wx0, w01 = wy0 * wx1;
            const float w10 = wy1 * wx0, w11 = wy1 * wx1;
#pragma unroll
            for (int cc = 0; cc < 8; ++cc) {
                const float* img =
                    h_prev + (size_t)(b * CH + g * CG + c0 + cc) * HW;
                float s = 0.0f;
                if (y0ok && x0ok) s += img[yi * WW + xi] * w00;
                if (y0ok && x1ok) s += img[yi * WW + xi + 1] * w01;
                if (y1ok && x0ok) s += img[(yi + 1) * WW + xi] * w10;
                if (y1ok && x1ok) s += img[(yi + 1) * WW + xi + 1] * w11;
                Bt[p * 32 + c0 + cc] = (__bf16)(s * mk);
            }
        }
#if HSA_ASYNC
        __builtin_amdgcn_s_wait_asynccnt(0);
#endif
        __syncthreads();

        // ---- WMMA: 8 waves x 4 N-tiles ------------------------------------
        {
            const v16bf a = load_frag_row32(&At[(wave * 16 + (lane & 15)) * 32], lane);
#pragma unroll
            for (int nt = 0; nt < 4; ++nt) {
                const v16bf bb =
                    load_frag_row32(&Bt[(nt * 16 + (lane & 15)) * 32], lane);
                acc[nt] = __builtin_amdgcn_wmma_f32_16x16x32_bf16(
                    false, a, false, bb, (short)0, acc[nt], false, false);
            }
        }
        __syncthreads();
    }

    // ---- epilogue: + b_dcn, store ----------------------------------------
    {
        const int n = lane & 15;
#pragma unroll
        for (int nt = 0; nt < 4; ++nt) {
            const int x = x0 + nt * 16 + n;
#pragma unroll
            for (int r = 0; r < 8; ++r) {
                const int o = wave * 16 + r + ((lane >= 16) ? 8 : 0);
                out[((size_t)(b * CH + o) * HH + h) * WW + x] = acc[nt][r] + b_dcn[o];
            }
        }
    }
}

// ===========================================================================
extern "C" void kernel_launch(void* const* d_in, const int* in_sizes, int n_in,
                              void* d_out, int out_size, void* d_ws, size_t ws_size,
                              hipStream_t stream) {
    const float* x_t    = (const float*)d_in[0];
    const float* h_prev = (const float*)d_in[1];
    const float* W_off  = (const float*)d_in[2];
    const float* b_off  = (const float*)d_in[3];
    const float* W_dcn  = (const float*)d_in[4];
    const float* b_dcn  = (const float*)d_in[5];
    float* out = (float*)d_out;

    const size_t S = (size_t)BATCH * G * KK * HH * WW;   // 2,359,296
    float*  dy = (float*)d_ws;
    float*  dx = dy + S;
    float*  mk = dx + S;
    __bf16* A1 = (__bf16*)(mk + S);                      // 112 x 2304 bf16
    __bf16* A2 = A1 + (size_t)OFFCP * K1TOT;             // 128 x 1152 bf16

    hsa_pack_weights<<<dim3(256), dim3(256), 0, stream>>>(W_off, W_dcn, A1, A2);

    const dim3 grid(BATCH * HH * (WW / 64));             // 1024 workgroups
    hsa_offset_conv<<<grid, dim3(256), 0, stream>>>(
        x_t, h_prev, A1, b_off, dy, dx, mk);
    hsa_dcn_sample_gemm<<<grid, dim3(256), 0, stream>>>(
        h_prev, A2, b_dcn, dy, dx, mk, out);
}
